// LinearEnsemble_26946624815474
// MI455X (gfx1250) — matrix operand order repaired and checked by reading, not compiled, not run on hardware
//
#include <hip/hip_runtime.h>
#include <stdint.h>

typedef __attribute__((ext_vector_type(16))) __bf16 v16bf;
typedef __attribute__((ext_vector_type(8)))  __bf16 v8bf;
typedef __attribute__((ext_vector_type(8)))  float  v8f;

#define B_TILE 128
#define O_TILE 128
#define K_TILE 32
#define LDSW   48   // padded LDS row stride in ushorts (96B, 32B-aligned rows)

// Split f32 -> bf16 hi (truncate) + bf16 lo (exact remainder, truncated),
// packed 4-wide, stored straight into the LDS staging tiles.
__device__ __forceinline__ void split_store(unsigned short* hi, unsigned short* lo, float4 v) {
  uint32_t u0 = __float_as_uint(v.x), u1 = __float_as_uint(v.y),
           u2 = __float_as_uint(v.z), u3 = __float_as_uint(v.w);
  float r0 = v.x - __uint_as_float(u0 & 0xFFFF0000u);
  float r1 = v.y - __uint_as_float(u1 & 0xFFFF0000u);
  float r2 = v.z - __uint_as_float(u2 & 0xFFFF0000u);
  float r3 = v.w - __uint_as_float(u3 & 0xFFFF0000u);
  uint2 h, l;
  h.x = (u0 >> 16) | (u1 & 0xFFFF0000u);
  h.y = (u2 >> 16) | (u3 & 0xFFFF0000u);
  l.x = (__float_as_uint(r0) >> 16) | (__float_as_uint(r1) & 0xFFFF0000u);
  l.y = (__float_as_uint(r2) >> 16) | (__float_as_uint(r3) & 0xFFFF0000u);
  *(uint2*)hi = h;
  *(uint2*)lo = l;
}

__global__ __launch_bounds__(256)
void le_bf16x3_gemm(const float* __restrict__ X,
                    const float* __restrict__ W,
                    const float* __restrict__ bias,
                    float* __restrict__ out)
{
  constexpr int I = 1024, O = 1024, M = 8;

  __shared__ unsigned short sXhi[B_TILE * LDSW];
  __shared__ unsigned short sXlo[B_TILE * LDSW];
  __shared__ unsigned short sWhi[O_TILE * LDSW];
  __shared__ unsigned short sWlo[O_TILE * LDSW];

  const int tid  = threadIdx.x;
  const int lane = tid & 31;
  const int wave = tid >> 5;     // 8 waves
  const int wm   = wave >> 1;    // 0..3 : 32-row group
  const int wn   = wave & 1;     // 0..1 : 64-col group

  const int rowBase = blockIdx.x * B_TILE;   // batch rows
  const int colBase = blockIdx.y * O_TILE;   // output cols
  const int m       = blockIdx.z;            // model index

  const float* Wm = W + (size_t)m * O * I;

  v8f acc[2][4];
#pragma unroll
  for (int i = 0; i < 2; ++i)
#pragma unroll
    for (int j = 0; j < 4; ++j)
      acc[i][j] = (v8f){0.f,0.f,0.f,0.f,0.f,0.f,0.f,0.f};

  const int lr = lane & 15;   // row (M) / col (N) within a 16-tile
  const int lh = lane >> 4;   // half-wave selector

  for (int kt = 0; kt < I; kt += K_TILE) {
    // ---- cooperative global loads (f32 tiles, 128x32 each) ----
    float4 xv[4], wv[4];
#pragma unroll
    for (int t = 0; t < 4; ++t) {
      int idx = tid + t * 256;             // 0..1023 over 128x8 float4s
      int r = idx >> 3, c4 = idx & 7;
      xv[t] = *(const float4*)(X + (size_t)(rowBase + r) * I + kt + c4 * 4);
      wv[t] = *(const float4*)(Wm + (size_t)(colBase + r) * I + kt + c4 * 4);
    }

    __syncthreads();   // previous iteration's LDS reads complete

    // ---- split f32 -> bf16 hi/lo, stage in LDS (once per block) ----
#pragma unroll
    for (int t = 0; t < 4; ++t) {
      int idx = tid + t * 256;
      int r = idx >> 3, c4 = idx & 7;
      split_store(sXhi + r * LDSW + c4 * 4, sXlo + r * LDSW + c4 * 4, xv[t]);
      split_store(sWhi + r * LDSW + c4 * 4, sWlo + r * LDSW + c4 * 4, wv[t]);
    }

    __syncthreads();   // tiles visible to all waves

    // ---- WMMA: bf16x3 (hi*hi + hi*lo + lo*hi), f32 accumulate ----
    // B fragments for the wave's 4 N-tiles (lane = N col; K contiguous)
    v16bf bh[4], bl[4];
#pragma unroll
    for (int nt = 0; nt < 4; ++nt) {
      const unsigned short* wh = sWhi + (wn * 64 + nt * 16 + lr) * LDSW + lh * 16;
      const unsigned short* wl = sWlo + (wn * 64 + nt * 16 + lr) * LDSW + lh * 16;
      bh[nt] = *(const v16bf*)wh;
      bl[nt] = *(const v16bf*)wl;
    }
#pragma unroll
    for (int mt = 0; mt < 2; ++mt) {
      const unsigned short* xh = sXhi + (wm * 32 + mt * 16 + lr) * LDSW;
      const unsigned short* xl = sXlo + (wm * 32 + mt * 16 + lr) * LDSW;
      // A fragment: lanes 0-15 hold K 0..15, lanes 16-31 hold K 16..31
      v8bf ah0 = *(const v8bf*)(xh + lh * 16);
      v8bf ah1 = *(const v8bf*)(xh + 8 + lh * 16);
      v8bf al0 = *(const v8bf*)(xl + lh * 16);
      v8bf al1 = *(const v8bf*)(xl + 8 + lh * 16);
      v16bf ah = __builtin_shufflevector(ah0, ah1, 0,1,2,3,4,5,6,7,8,9,10,11,12,13,14,15);
      v16bf al = __builtin_shufflevector(al0, al1, 0,1,2,3,4,5,6,7,8,9,10,11,12,13,14,15);
#pragma unroll
      for (int nt = 0; nt < 4; ++nt) {
        acc[mt][nt] = __builtin_amdgcn_wmma_f32_16x16x32_bf16(
            false, ah, false, bh[nt], (short)0, acc[mt][nt], false, false);
        acc[mt][nt] = __builtin_amdgcn_wmma_f32_16x16x32_bf16(
            false, ah, false, bl[nt], (short)0, acc[mt][nt], false, false);
        acc[mt][nt] = __builtin_amdgcn_wmma_f32_16x16x32_bf16(
            false, al, false, bh[nt], (short)0, acc[mt][nt], false, false);
      }
    }
  }

  // ---- epilogue: bias add + store (out is [B, M, O]) ----
#pragma unroll
  for (int mt = 0; mt < 2; ++mt) {
#pragma unroll
    for (int nt = 0; nt < 4; ++nt) {
      int o  = colBase + wn * 64 + nt * 16 + lr;
      float bv = bias[m * O + o];
      int rb = rowBase + wm * 32 + mt * 16 + lh * 8;   // C: VGPR r -> row r / r+8
      float* op = out + (size_t)rb * (size_t)(M * O) + (size_t)m * O + o;
#pragma unroll
      for (int r = 0; r < 8; ++r)
        op[(size_t)r * (M * O)] = acc[mt][nt][r] + bv;
    }
  }
}

extern "C" void kernel_launch(void* const* d_in, const int* in_sizes, int n_in,
                              void* d_out, int out_size, void* d_ws, size_t ws_size,
                              hipStream_t stream) {
  const float* x = (const float*)d_in[0];   // [B, 1024]
  const float* w = (const float*)d_in[1];   // [8, 1024, 1024]
  const float* b = (const float*)d_in[2];   // [8, 1024]
  float* out = (float*)d_out;               // [B, 8, 1024]

  const int I = 1024, O = 1024, M = 8;
  int B = in_sizes[0] / I;                  // 4096

  dim3 grid(B / B_TILE, O / O_TILE, M);     // (32, 8, 8)
  dim3 block(256);
  le_bf16x3_gemm<<<grid, block, 0, stream>>>(x, w, b, out);
}
